// HAFR_15599321219495
// MI455X (gfx1250) — matrix operand order repaired and checked by reading, not compiled, not run on hardware
//
#include <hip/hip_runtime.h>
#include <hip/hip_bf16.h>

// ---------------------------------------------------------------------------
// CDNA5 (gfx1250) WMMA types
// ---------------------------------------------------------------------------
typedef __bf16 bf16_t;
typedef __attribute__((ext_vector_type(16))) __bf16 v16bf;
typedef __attribute__((ext_vector_type(8)))  float  v8f;

#define HB 16384   // batch
#define HN 20      // ingredients
#define HD 64      // dim
#define HIMG 2048

// Toggle for the CDNA5 async global->LDS staging path (inline asm).
#define USE_ASYNC_LDS 1

__device__ __forceinline__ float rowReduce16(float x) {
    x += __shfl_xor(x, 1, 16);
    x += __shfl_xor(x, 2, 16);
    x += __shfl_xor(x, 4, 16);
    x += __shfl_xor(x, 8, 16);
    return x;
}

#if USE_ASYNC_LDS
// GLOBAL_LOAD_ASYNC_TO_LDS_B128: copies 16 B from global memory into LDS at vdst, tracked by ASYNCcnt
// vdst carries the workgroup-relative LDS byte offset (== AS(3) pointer value,
// i.e. the low 32 bits of the generic shared pointer).
__device__ __forceinline__ void async_load_b128(unsigned int lds_off, const void* gptr) {
    asm volatile("global_load_async_to_lds_b128 %0, %1, off"
                 :: "v"(lds_off), "v"((unsigned long long)(uintptr_t)gptr)
                 : "memory");
}
__device__ __forceinline__ void wait_async0() {
    asm volatile("s_wait_asynccnt 0x0" ::: "memory");
}
#endif

// ---------------------------------------------------------------------------
// 1) Repack a [K,64] f32 weight matrix into bf16 B-fragment order:
//    dst[((kc*4+nt)*32+lane)*16 + e] = W[(kc*32 + (e>>3)*16 + (lane>>4)*8 + (e&7))*64 + nt*16 + (lane&15)]
// ---------------------------------------------------------------------------
__global__ void prep_frag(const float* __restrict__ W, bf16_t* __restrict__ dst, int nkc) {
    int total = nkc * 2048; // 4 ntiles * 32 lanes * 16 elems per k-chunk
    for (int idx = blockIdx.x * blockDim.x + threadIdx.x; idx < total;
         idx += gridDim.x * blockDim.x) {
        int e    = idx & 15;
        int lane = (idx >> 4) & 31;
        int nt   = (idx >> 9) & 3;
        int kc   = idx >> 11;
        int K    = kc * 32 + (e >> 3) * 16 + (lane >> 4) * 8 + (e & 7);
        int Nc   = nt * 16 + (lane & 15);
        dst[idx] = (bf16_t)W[K * 64 + Nc];
    }
}

// ---------------------------------------------------------------------------
// 2) Gather user / item embeddings
// ---------------------------------------------------------------------------
__global__ void gather_emb(const int* __restrict__ ui, const int* __restrict__ ii,
                           const float* __restrict__ uemb, const float* __restrict__ iemb,
                           float* __restrict__ ue, float* __restrict__ ie) {
    int total = HB * HD;
    for (int idx = blockIdx.x * blockDim.x + threadIdx.x; idx < total;
         idx += gridDim.x * blockDim.x) {
        int b = idx >> 6, d = idx & 63;
        ue[idx] = uemb[(size_t)ui[b] * HD + d];
        ie[idx] = iemb[(size_t)ii[b] * HD + d];
    }
}

// ---------------------------------------------------------------------------
// 3) img = image_input @ W_image (bf16 WMMA, f32 accum) + biases.
//    Memory-bound (128 MB A stream): block = 8 waves = 2 tiles x 4-way K-split
//    (each wave does 16 of 64 K-chunks), partials reduced through LDS.
//    512 blocks * 8 waves = 4096 in-flight waves for latency hiding.
// ---------------------------------------------------------------------------
__global__ void __launch_bounds__(256)
image_gemm(const float* __restrict__ X,       // [B,2048]
           const bf16_t* __restrict__ fragW,  // [64][4][32][16] bf16
           const float* __restrict__ b1, const float* __restrict__ b2,
           float* __restrict__ Y) {            // [B,64]
    __shared__ float smAcc[2][4][16][64];      // 32 KB
    const int tid  = threadIdx.x;
    const int lane = tid & 31;
    const int wave = tid >> 5;
    const int tl   = wave >> 2;     // tile within block
    const int ksub = wave & 3;      // K-split slot
    const int tile = blockIdx.x * 2 + tl;
    const int half = lane >> 4;
    const int l16  = lane & 15;
    const float* Arow = X + (size_t)(tile * 16 + l16) * HIMG;

    v8f acc[4] = {};
    for (int k = 0; k < 16; ++k) {
        const int kc = ksub * 16 + k;
        const int kb = kc * 32;
        const float4* p0 = reinterpret_cast<const float4*>(Arow + kb + half * 8);
        const float4* p1 = reinterpret_cast<const float4*>(Arow + kb + 16 + half * 8);
        float4 x0 = p0[0], x1 = p0[1];
        float4 x2 = p1[0], x3 = p1[1];
        v16bf a;
        a[0]=(bf16_t)x0.x;  a[1]=(bf16_t)x0.y;  a[2]=(bf16_t)x0.z;  a[3]=(bf16_t)x0.w;
        a[4]=(bf16_t)x1.x;  a[5]=(bf16_t)x1.y;  a[6]=(bf16_t)x1.z;  a[7]=(bf16_t)x1.w;
        a[8]=(bf16_t)x2.x;  a[9]=(bf16_t)x2.y;  a[10]=(bf16_t)x2.z; a[11]=(bf16_t)x2.w;
        a[12]=(bf16_t)x3.x; a[13]=(bf16_t)x3.y; a[14]=(bf16_t)x3.z; a[15]=(bf16_t)x3.w;
        #pragma unroll
        for (int nt = 0; nt < 4; ++nt) {
            v16bf bfr = *reinterpret_cast<const v16bf*>(
                fragW + ((size_t)(kc * 4 + nt) * 32 + lane) * 16);
            acc[nt] = __builtin_amdgcn_wmma_f32_16x16x32_bf16(
                false, a, false, bfr, (short)0, acc[nt], false, false);
        }
    }
    #pragma unroll
    for (int nt = 0; nt < 4; ++nt)
        #pragma unroll
        for (int i = 0; i < 8; ++i)
            smAcc[tl][ksub][half * 8 + i][nt * 16 + l16] = acc[nt][i];
    __syncthreads();

    for (int idx = tid; idx < 2 * 16 * 64; idx += blockDim.x) {
        int t = idx >> 10;
        int r = (idx >> 6) & 15;
        int c = idx & 63;
        float sum = smAcc[t][0][r][c] + smAcc[t][1][r][c] +
                    smAcc[t][2][r][c] + smAcc[t][3][r][c];
        Y[(size_t)((blockIdx.x * 2 + t) * 16 + r) * HD + c] = sum + b1[c] + b2[c];
    }
}

// ---------------------------------------------------------------------------
// 4) Ingredient-level attention, fully fused.
//    block = 4 batch rows * 20 ingredients = 80 A-rows = 5 wave tiles (160 thr)
//    qe rows staged global->LDS with async-to-LDS (ASYNCcnt), then bf16-ized.
// ---------------------------------------------------------------------------
__global__ void __launch_bounds__(160)
ingre_attn(const int* __restrict__ ingre_input, const int* __restrict__ ingre_num,
           const float* __restrict__ ingre_emb,
           const float* __restrict__ ue_ws, const float* __restrict__ img_ws,
           const bf16_t* __restrict__ fragW,     // [6][4][32][16]
           const float* __restrict__ wb, const float* __restrict__ bb,
           const float* __restrict__ vvec,
           float* __restrict__ out_ia) {         // [B,64]
    __shared__ bf16_t smA[80 * 192];   // concat rows, bf16
    __shared__ float  smQ[80 * 64];    // qe f32 for the weighted sum
    __shared__ float  smS[80];         // per-row attention scores

    const int tid = threadIdx.x;
    const int b0  = blockIdx.x * 4;

#if USE_ASYNC_LDS
    // phase 1a: async global->LDS gather of qe rows (16B quads, all lanes active)
    for (int idx = tid; idx < 80 * 16; idx += blockDim.x) {
        int r = idx >> 4, qd = idx & 15;
        int lb = r / HN, j = r - lb * HN;
        int gb = b0 + lb;
        const float* src = ingre_emb + (size_t)ingre_input[gb * HN + j] * HD + qd * 4;
        async_load_b128((unsigned int)(uintptr_t)&smQ[r * 64 + qd * 4], src);
    }
    // phase 1b: ue / img halves of the concat rows (bf16) while asyncs fly
    for (int idx = tid; idx < 80 * 64; idx += blockDim.x) {
        int r = idx >> 6, d = idx & 63;
        int gb = b0 + r / HN;
        smA[r * 192 + 64 + d]  = (bf16_t)ue_ws[gb * HD + d];
        smA[r * 192 + 128 + d] = (bf16_t)img_ws[gb * HD + d];
    }
    wait_async0();
    __syncthreads();
    // phase 2: bf16-ize qe out of LDS into the A-operand
    for (int idx = tid; idx < 80 * 64; idx += blockDim.x) {
        int r = idx >> 6, d = idx & 63;
        smA[r * 192 + d] = (bf16_t)smQ[r * 64 + d];
    }
    __syncthreads();
#else
    for (int idx = tid; idx < 80 * 64; idx += blockDim.x) {
        int r = idx >> 6, d = idx & 63;
        int lb = r / HN, j = r - lb * HN;
        int gb = b0 + lb;
        float q  = ingre_emb[(size_t)ingre_input[gb * HN + j] * HD + d];
        smQ[idx] = q;
        smA[r * 192 + d]        = (bf16_t)q;
        smA[r * 192 + 64 + d]   = (bf16_t)ue_ws[gb * HD + d];
        smA[r * 192 + 128 + d]  = (bf16_t)img_ws[gb * HD + d];
    }
    __syncthreads();
#endif

    const int lane = tid & 31, wave = tid >> 5;
    const int half = lane >> 4, l16 = lane & 15;
    const int rloc = wave * 16 + l16;

    v8f acc[4] = {};
    #pragma unroll
    for (int kc = 0; kc < 6; ++kc) {
        v16bf a;
        const bf16_t* base = smA + rloc * 192 + kc * 32 + half * 8;
        #pragma unroll
        for (int e = 0; e < 8; ++e) a[e] = base[e];
        #pragma unroll
        for (int e = 0; e < 8; ++e) a[8 + e] = base[16 + e];
        #pragma unroll
        for (int nt = 0; nt < 4; ++nt) {
            v16bf bfr = *reinterpret_cast<const v16bf*>(
                fragW + ((size_t)(kc * 4 + nt) * 32 + lane) * 16);
            acc[nt] = __builtin_amdgcn_wmma_f32_16x16x32_bf16(
                false, a, false, bfr, (short)0, acc[nt], false, false);
        }
    }
    // fused tanh + dot(v) epilogue
    float s[8];
    #pragma unroll
    for (int i = 0; i < 8; ++i) s[i] = 0.f;
    #pragma unroll
    for (int nt = 0; nt < 4; ++nt) {
        int col = nt * 16 + l16;
        float bias = wb[col] + bb[col];
        float vv = vvec[col];
        #pragma unroll
        for (int i = 0; i < 8; ++i)
            s[i] += tanhf(acc[nt][i] + bias) * vv;
    }
    #pragma unroll
    for (int i = 0; i < 8; ++i) {
        float red = rowReduce16(s[i]);
        if (l16 == 0) smS[wave * 16 + half * 8 + i] = red;
    }
    __syncthreads();

    // masked softmax over N=20 per batch row (4 rows, serial -> tiny)
    if (tid < 4) {
        int num = ingre_num[b0 + tid];
        float mx = -3.0e38f;
        for (int j = 0; j < HN; ++j) {
            float val = (j < num) ? smS[tid * HN + j] : -1.0e12f;
            smS[tid * HN + j] = val;
            mx = fmaxf(mx, val);
        }
        float sum = 0.f;
        for (int j = 0; j < HN; ++j) {
            float e = __expf(smS[tid * HN + j] - mx);
            smS[tid * HN + j] = e;
            sum += e;
        }
        float inv = 1.f / sum;
        for (int j = 0; j < HN; ++j) smS[tid * HN + j] *= inv;
    }
    __syncthreads();

    // weighted sum of qe
    for (int idx = tid; idx < 4 * 64; idx += blockDim.x) {
        int lb = idx >> 6, d = idx & 63;
        float a = 0.f;
        #pragma unroll
        for (int j = 0; j < HN; ++j)
            a += smS[lb * HN + j] * smQ[(lb * HN + j) * 64 + d];
        out_ia[(size_t)(b0 + lb) * HD + d] = a;
    }
}

// ---------------------------------------------------------------------------
// 5) Component scores in the reference's flat concat order: sc[comp*B + b]
//    block = 16 batch rows, 3 waves (one per component), K=128
// ---------------------------------------------------------------------------
__global__ void __launch_bounds__(96)
comp_scores(const float* __restrict__ ue_ws, const float* __restrict__ ie_ws,
            const float* __restrict__ ia_ws, const float* __restrict__ img_ws,
            const bf16_t* __restrict__ fragW,  // [4][4][32][16]
            const float* __restrict__ wb, const float* __restrict__ bb,
            const float* __restrict__ vc,
            float* __restrict__ sc_ws) {        // [3*B]
    __shared__ bf16_t smCP[48 * 128];
    const int tid = threadIdx.x;
    const int b0  = blockIdx.x * 16;

    for (int idx = tid; idx < 48 * 128; idx += blockDim.x) {
        int r = idx >> 7, c = idx & 127;
        int comp = r >> 4, lb = r & 15;
        int gb = b0 + lb;
        float val;
        if (c < 64) val = ue_ws[gb * HD + c];
        else {
            int d = c - 64;
            val = (comp == 0) ? ie_ws[gb * HD + d]
                : (comp == 1) ? ia_ws[gb * HD + d]
                              : img_ws[gb * HD + d];
        }
        smCP[idx] = (bf16_t)val;
    }
    __syncthreads();

    const int lane = tid & 31, wave = tid >> 5;
    const int half = lane >> 4, l16 = lane & 15;
    const int comp = wave;

    v8f acc[4] = {};
    #pragma unroll
    for (int kc = 0; kc < 4; ++kc) {
        v16bf a;
        const bf16_t* base = smCP + (comp * 16 + l16) * 128 + kc * 32 + half * 8;
        #pragma unroll
        for (int e = 0; e < 8; ++e) a[e] = base[e];
        #pragma unroll
        for (int e = 0; e < 8; ++e) a[8 + e] = base[16 + e];
        #pragma unroll
        for (int nt = 0; nt < 4; ++nt) {
            v16bf bfr = *reinterpret_cast<const v16bf*>(
                fragW + ((size_t)(kc * 4 + nt) * 32 + lane) * 16);
            acc[nt] = __builtin_amdgcn_wmma_f32_16x16x32_bf16(
                false, a, false, bfr, (short)0, acc[nt], false, false);
        }
    }
    float s[8];
    #pragma unroll
    for (int i = 0; i < 8; ++i) s[i] = 0.f;
    #pragma unroll
    for (int nt = 0; nt < 4; ++nt) {
        int col = nt * 16 + l16;
        float bias = wb[col] + bb[col];
        float vv = vc[col];
        #pragma unroll
        for (int i = 0; i < 8; ++i)
            s[i] += tanhf(acc[nt][i] + bias) * vv;
    }
    #pragma unroll
    for (int i = 0; i < 8; ++i) {
        float red = rowReduce16(s[i]);
        if (l16 == 0)
            sc_ws[(size_t)comp * HB + b0 + half * 8 + i] = red;
    }
}

// ---------------------------------------------------------------------------
// 6) Scrambled component softmax + item_att + prediction head
//    reference quirk: scores[i,j] = flat[3i+j]  (comp = (3i+j)/B, b = (3i+j)%B)
//    block = 16 batch rows, 4 waves; wave0 runs the 16x64 K=192 relu-GEMM
// ---------------------------------------------------------------------------
__global__ void __launch_bounds__(128)
head_kernel(const float* __restrict__ ue_ws, const float* __restrict__ ie_ws,
            const float* __restrict__ ia_ws, const float* __restrict__ img_ws,
            const float* __restrict__ sc_ws,
            const bf16_t* __restrict__ fragCat, // [6][4][32][16]
            const float* __restrict__ wb, const float* __restrict__ bb,
            const float* __restrict__ h_w, const float* __restrict__ h_b,
            float* __restrict__ out) {           // [B]
    __shared__ bf16_t smUIC[16 * 192];
    const int tid = threadIdx.x;
    const int b0  = blockIdx.x * 16;

    for (int idx = tid; idx < 16 * 64; idx += blockDim.x) {
        int lb = idx >> 6, d = idx & 63;
        int gb = b0 + lb;
        float s0 = sc_ws[(size_t)3 * gb + 0];
        float s1 = sc_ws[(size_t)3 * gb + 1];
        float s2 = sc_ws[(size_t)3 * gb + 2];
        float mx = fmaxf(s0, fmaxf(s1, s2));
        float e0 = __expf(s0 - mx), e1 = __expf(s1 - mx), e2 = __expf(s2 - mx);
        float inv = 1.f / (e0 + e1 + e2);
        float ia = (e0 * ie_ws[gb * HD + d] + e1 * ia_ws[gb * HD + d] +
                    e2 * img_ws[gb * HD + d]) * inv;
        float u = ue_ws[gb * HD + d];
        smUIC[lb * 192 + d]       = (bf16_t)u;
        smUIC[lb * 192 + 64 + d]  = (bf16_t)ia;
        smUIC[lb * 192 + 128 + d] = (bf16_t)(u * ia);
    }
    __syncthreads();

    const int lane = tid & 31, wave = tid >> 5;
    if (wave == 0) {
        const int half = lane >> 4, l16 = lane & 15;
        v8f acc[4] = {};
        #pragma unroll
        for (int kc = 0; kc < 6; ++kc) {
            v16bf a;
            const bf16_t* base = smUIC + l16 * 192 + kc * 32 + half * 8;
            #pragma unroll
            for (int e = 0; e < 8; ++e) a[e] = base[e];
            #pragma unroll
            for (int e = 0; e < 8; ++e) a[8 + e] = base[16 + e];
            #pragma unroll
            for (int nt = 0; nt < 4; ++nt) {
                v16bf bfr = *reinterpret_cast<const v16bf*>(
                    fragCat + ((size_t)(kc * 4 + nt) * 32 + lane) * 16);
                acc[nt] = __builtin_amdgcn_wmma_f32_16x16x32_bf16(
                    false, a, false, bfr, (short)0, acc[nt], false, false);
            }
        }
        float s[8];
        #pragma unroll
        for (int i = 0; i < 8; ++i) s[i] = 0.f;
        #pragma unroll
        for (int nt = 0; nt < 4; ++nt) {
            int col = nt * 16 + l16;
            float bias = wb[col] + bb[col];
            float hw = h_w[col];
            #pragma unroll
            for (int i = 0; i < 8; ++i)
                s[i] += fmaxf(acc[nt][i] + bias, 0.f) * hw;
        }
        float hb = h_b[0];
        #pragma unroll
        for (int i = 0; i < 8; ++i) {
            float red = rowReduce16(s[i]);
            if (l16 == 0) out[b0 + half * 8 + i] = red + hb;
        }
    }
}

// ---------------------------------------------------------------------------
extern "C" void kernel_launch(void* const* d_in, const int* in_sizes, int n_in,
                              void* d_out, int out_size, void* d_ws, size_t ws_size,
                              hipStream_t stream) {
    const int*   user_input    = (const int*)d_in[0];
    const int*   item_input    = (const int*)d_in[1];
    const int*   ingre_input   = (const int*)d_in[2];
    const float* image_input   = (const float*)d_in[3];
    const int*   ingre_num     = (const int*)d_in[4];
    const float* user_emb      = (const float*)d_in[5];
    const float* item_emb      = (const float*)d_in[6];
    const float* ingre_emb     = (const float*)d_in[7];
    const float* W_image_w     = (const float*)d_in[8];
    const float* W_image_b     = (const float*)d_in[9];
    const float* b_image       = (const float*)d_in[10];
    const float* W_concat_w    = (const float*)d_in[11];
    const float* W_concat_b    = (const float*)d_in[12];
    const float* b_concat      = (const float*)d_in[13];
    const float* h_w           = (const float*)d_in[14];
    const float* h_b           = (const float*)d_in[15];
    const float* W_att_ingre_w = (const float*)d_in[16];
    const float* W_att_ingre_b = (const float*)d_in[17];
    const float* b_att_ingre   = (const float*)d_in[18];
    const float* v             = (const float*)d_in[19];
    const float* W_att_com_w   = (const float*)d_in[20];
    const float* W_att_com_b   = (const float*)d_in[21];
    const float* b_att_com     = (const float*)d_in[22];
    const float* v_c           = (const float*)d_in[23];

    const size_t SZ_BD = (size_t)HB * HD * sizeof(float);   // 4 MiB
    const size_t SZ_SC = (size_t)3 * HB * sizeof(float);
    char* ws = (char*)d_ws;
    float* img_ws = (float*)(ws);
    float* ue_ws  = (float*)(ws + SZ_BD);
    float* ie_ws  = (float*)(ws + 2 * SZ_BD);
    float* ia_ws  = (float*)(ws + 3 * SZ_BD);
    float* sc_ws  = (float*)(ws + 4 * SZ_BD);
    bf16_t* fragImg = (bf16_t*)(ws + 4 * SZ_BD + SZ_SC);
    bf16_t* fragIng = fragImg + (size_t)HIMG * 64;
    bf16_t* fragCom = fragIng + (size_t)192 * 64;
    bf16_t* fragCat = fragCom + (size_t)128 * 64;

    prep_frag<<<512, 256, 0, stream>>>(W_image_w,     fragImg, 64);
    prep_frag<<<48,  256, 0, stream>>>(W_att_ingre_w, fragIng, 6);
    prep_frag<<<32,  256, 0, stream>>>(W_att_com_w,   fragCom, 4);
    prep_frag<<<48,  256, 0, stream>>>(W_concat_w,    fragCat, 6);

    gather_emb<<<2048, 256, 0, stream>>>(user_input, item_input, user_emb, item_emb,
                                         ue_ws, ie_ws);

    image_gemm<<<HB / 32, 256, 0, stream>>>(image_input, fragImg, W_image_b, b_image,
                                            img_ws);

    ingre_attn<<<HB / 4, 160, 0, stream>>>(ingre_input, ingre_num, ingre_emb,
                                           ue_ws, img_ws, fragIng,
                                           W_att_ingre_b, b_att_ingre, v, ia_ws);

    comp_scores<<<HB / 16, 96, 0, stream>>>(ue_ws, ie_ws, ia_ws, img_ws, fragCom,
                                            W_att_com_b, b_att_com, v_c, sc_ws);

    head_kernel<<<HB / 16, 128, 0, stream>>>(ue_ws, ie_ws, ia_ws, img_ws, sc_ws,
                                             fragCat, W_concat_b, b_concat,
                                             h_w, h_b, (float*)d_out);
}
